// DTATTransformer_14757507629069
// MI455X (gfx1250) — matrix operand order repaired and checked by reading, not compiled, not run on hardware
//
#include <hip/hip_runtime.h>
#include <hip/hip_bf16.h>

#define H   16
#define DH  128
#define CS  64
#define TOPK 32
#define B_  2
#define T_  2048
#define C_  2048
#define NCH 32            // T_/CS
#define M_  (B_*T_)       // 4096

typedef __attribute__((ext_vector_type(16))) __bf16 v16bf;
typedef __attribute__((ext_vector_type(8)))  float  v8f;

static __device__ __forceinline__ v16bf mkfrag(uint4 lo, uint4 hi) {
    union { uint4 u[2]; v16bf v; } x;
    x.u[0] = lo; x.u[1] = hi;
    return x.v;
}

static __device__ __forceinline__ float n2n(float x) {
    if (__builtin_isnan(x)) return 0.f;
    if (__builtin_isinf(x)) return x > 0.f ? 1.f : -1.f;
    return x;
}

// async global -> LDS copy of 16 bytes (ASYNCcnt-tracked, CDNA5)
static __device__ __forceinline__ void async_copy_b128(const void* gaddr, void* lds) {
    unsigned l32 = (unsigned)(unsigned long long)lds;   // low 32 bits of generic ptr = LDS offset
    asm volatile("global_load_async_to_lds_b128 %0, %1, off"
                 :: "v"(l32), "v"(gaddr) : "memory");
}

static __device__ __forceinline__ void wait_async0() {
#if __has_builtin(__builtin_amdgcn_s_wait_asynccnt)
    __builtin_amdgcn_s_wait_asynccnt(0);
#else
    asm volatile("s_wait_asynccnt 0x0" ::: "memory");
#endif
}

// ---------------------------------------------------------------- cast f32->bf16
__global__ void __launch_bounds__(256) cast_bf16_kernel(
    const float* __restrict__ in, __bf16* __restrict__ out, int n) {
    int i = blockIdx.x * blockDim.x + threadIdx.x;
    int stride = gridDim.x * blockDim.x;
    for (; i < n; i += stride) out[i] = (__bf16)in[i];
}

// ------------------------------------------- transpose+cast: WT[n][k] = W[k][n]
__global__ void __launch_bounds__(256) transpose_cast_kernel(
    const float* __restrict__ W, __bf16* __restrict__ WT) {
    __shared__ float tile[32][33];
    int n0 = blockIdx.x * 32;
    int k0 = blockIdx.y * 32;
    int tx = threadIdx.x & 31;
    int ty = threadIdx.x >> 5;                       // 0..7
    for (int r = ty; r < 32; r += 8)
        tile[r][tx] = W[(k0 + r) * C_ + n0 + tx];
    __syncthreads();
    for (int r = ty; r < 32; r += 8)
        WT[(n0 + r) * C_ + k0 + tx] = (__bf16)tile[tx][r];
}

// ---------------------------------------------------------------- bf16 WMMA GEMM
// Out[M_ x C_] = A[M_ x C_] @ B (given as BT[n][k]) + bias
// Double-buffered async global->LDS staging.
template<bool OUT_BF16, bool NAN2NUM>
__global__ void __launch_bounds__(256) gemm_bf16_kernel(
    const __bf16* __restrict__ A,
    const __bf16* __restrict__ BT,
    const float*  __restrict__ bias,
    void* __restrict__ Out) {
    const int LDA = 40;                              // halfs/row (80B, 16B-aligned)
    __shared__ __bf16 As[2][128 * LDA];
    __shared__ __bf16 Bs[2][128 * LDA];

    int tid  = threadIdx.x;
    int lane = tid & 31, wave = tid >> 5;
    int wr = wave & 3;                               // 0..3 -> 32-row block
    int wc = wave >> 2;                              // 0..1 -> 64-col block
    int lr = lane & 15;
    int rowBase = blockIdx.x * 128;
    int colBase = blockIdx.y * 128;

    v8f acc[2][4];
    #pragma unroll
    for (int i = 0; i < 2; ++i)
        #pragma unroll
        for (int j = 0; j < 4; ++j) acc[i][j] = {};

    const int koA = (lane < 16) ? 0 : 8;
    const int koB = (lane < 16) ? 0 : 8;

    // per-thread staging coordinates: 2 x 16B segments for A and for B
    int r0 = tid >> 2,        c80 = (tid & 3) * 8;
    int r1 = (tid + 256) >> 2, c81 = ((tid + 256) & 3) * 8;

    auto issue = [&](int buf, int k0) {
        async_copy_b128(&A [(rowBase + r0) * C_ + k0 + c80], &As[buf][r0 * LDA + c80]);
        async_copy_b128(&BT[(colBase + r0) * C_ + k0 + c80], &Bs[buf][r0 * LDA + c80]);
        async_copy_b128(&A [(rowBase + r1) * C_ + k0 + c81], &As[buf][r1 * LDA + c81]);
        async_copy_b128(&BT[(colBase + r1) * C_ + k0 + c81], &Bs[buf][r1 * LDA + c81]);
    };

    issue(0, 0);

    const int NK = C_ / 32;                          // 64 K-steps
    for (int ks = 0; ks < NK; ++ks) {
        int buf = ks & 1;
        wait_async0();
        __syncthreads();                             // staged tile visible; prior reads done
        if (ks + 1 < NK) issue(buf ^ 1, (ks + 1) * 32);

        v16bf af[2], bfr[4];
        #pragma unroll
        for (int i = 0; i < 2; ++i) {
            int r = wr * 32 + i * 16 + lr;
            af[i] = mkfrag(*(const uint4*)&As[buf][r * LDA + koA],
                           *(const uint4*)&As[buf][r * LDA + koA + 16]);
        }
        #pragma unroll
        for (int j = 0; j < 4; ++j) {
            int n = wc * 64 + j * 16 + lr;
            bfr[j] = mkfrag(*(const uint4*)&Bs[buf][n * LDA + koB],
                            *(const uint4*)&Bs[buf][n * LDA + koB + 16]);
        }
        #pragma unroll
        for (int i = 0; i < 2; ++i)
            #pragma unroll
            for (int j = 0; j < 4; ++j)
                acc[i][j] = __builtin_amdgcn_wmma_f32_16x16x32_bf16(
                    false, af[i], false, bfr[j], (short)0, acc[i][j], false, false);
    }

    #pragma unroll
    for (int i = 0; i < 2; ++i) {
        #pragma unroll
        for (int j = 0; j < 4; ++j) {
            int col = colBase + wc * 64 + j * 16 + lr;
            float bv = bias[col];
            #pragma unroll
            for (int r = 0; r < 8; ++r) {
                int row = rowBase + wr * 32 + i * 16 + ((lane < 16) ? r : r + 8);
                float v = acc[i][j][r] + bv;
                if (NAN2NUM) v = n2n(v);
                if (OUT_BF16) ((__bf16*)Out)[row * C_ + col] = (__bf16)v;
                else          ((float*) Out)[row * C_ + col] = v;
            }
        }
    }
}

// ------------------------------------------------------------- sparse attention
// one block per (b, h, q-chunk n); 512 threads = 16 waves
__global__ void __launch_bounds__(512) attn_kernel(
    const __bf16* __restrict__ q,     // [M_, C_]
    const __bf16* __restrict__ k,
    const __bf16* __restrict__ v,
    const float*  __restrict__ imp,   // [B_, T_, H]
    const float*  __restrict__ temps, // [B_, H]
    __bf16* __restrict__ attn) {      // [M_, C_]
    const int LK = 136;               // Ks pitch (272B)
    const int LV = 72;                // Vt pitch (144B)
    const int LS = 68;                // Ss pitch (f32, 272B)
    const int LP = 72;                // Ps pitch (144B)
    __shared__ __bf16 Ks[64 * LK];
    __shared__ __bf16 Vt[128 * LV];
    __shared__ float  Ss[64 * LS];
    __shared__ __bf16 Ps[64 * LP];
    __shared__ float  mw[64];
    __shared__ float  nrm[64];

    int blk = blockIdx.x;
    int n = blk & (NCH - 1);
    int h = (blk >> 5) & (H - 1);
    int b = blk >> 9;

    int tid = threadIdx.x;
    int lane = tid & 31, wave = tid >> 5;
    int lr = lane & 15;
    int rq = wave >> 2;               // 0..3: 16-row block of q chunk
    int ce = wave & 3;                // 0..3: 16-col block of score tile

    float tclip = fminf(fmaxf(temps[b * H + h], 0.1f), 100.f);
    float sc = 0.08838834764831845f / tclip;   // 1/sqrt(128)/temp

    if (tid < 64) {
        float is = imp[(b * T_ + n * CS + tid) * H + h];
        float s1 = 1.f / (1.f + __expf(-is));
        mw[tid]  = 1.f / (1.f + __expf(-(s1 - 0.5f) * 10.f));
        nrm[tid] = 0.f;
    }

    // preload q A-fragments (4 K-steps over D=128)
    v16bf qa[4];
    {
        int row  = b * T_ + n * CS + rq * 16 + lr;
        int ko   = (lane < 16) ? 0 : 8;
        const __bf16* base = &q[row * C_ + h * DH];
        #pragma unroll
        for (int kk = 0; kk < 4; ++kk) {
            const __bf16* p = base + kk * 32 + ko;
            qa[kk] = mkfrag(*(const uint4*)p, *(const uint4*)(p + 16));
        }
    }
    __syncthreads();

    v8f oacc[2];
    oacc[0] = {}; oacc[1] = {};

    for (int m = 0; m < NCH; ++m) {
        // stage k chunk row-major (async -> LDS), v chunk transposed (via VGPRs)
        for (int s = tid; s < 1024; s += 512) {
            int e  = s >> 4;
            int d8 = (s & 15) * 8;
            int row = b * T_ + m * CS + e;
            async_copy_b128(&k[row * C_ + h * DH + d8], &Ks[e * LK + d8]);
            uint4 vq = *(const uint4*)&v[row * C_ + h * DH + d8];
            const __bf16* vp = (const __bf16*)&vq;
            #pragma unroll
            for (int j = 0; j < 8; ++j) Vt[(d8 + j) * LV + e] = vp[j];
        }
        wait_async0();
        __syncthreads();

        // 16x16 score tile per wave: scores = q @ k^T
        v8f s8 = {};
        {
            int koB = (lane < 16) ? 0 : 8;
            int e = ce * 16 + lr;
            #pragma unroll
            for (int kk = 0; kk < 4; ++kk) {
                v16bf bfr = mkfrag(*(const uint4*)&Ks[e * LK + kk * 32 + koB],
                                   *(const uint4*)&Ks[e * LK + kk * 32 + koB + 16]);
                s8 = __builtin_amdgcn_wmma_f32_16x16x32_bf16(
                    false, qa[kk], false, bfr, (short)0, s8, false, false);
            }
        }
        #pragma unroll
        for (int r = 0; r < 8; ++r) {
            int c = rq * 16 + ((lane < 16) ? r : r + 8);
            Ss[c * LS + ce * 16 + lr] = s8[r] * sc * mw[c];
        }
        __syncthreads();

        // top-k: keep element iff (#strictly greater in its 64-row) < TOPK
        float sv[8];
        {
            int c = tid >> 3, g = tid & 7;
            int cnt[8];
            #pragma unroll
            for (int j = 0; j < 8; ++j) { sv[j] = Ss[c * LS + g * 8 + j]; cnt[j] = 0; }
            for (int e2 = 0; e2 < 64; ++e2) {
                float o = Ss[c * LS + e2];
                #pragma unroll
                for (int j = 0; j < 8; ++j) cnt[j] += (o > sv[j]) ? 1 : 0;
            }
            #pragma unroll
            for (int j = 0; j < 8; ++j) if (cnt[j] >= TOPK) sv[j] = 0.f;
        }
        __syncthreads();
        {
            int c = tid >> 3, g = tid & 7;
            #pragma unroll
            for (int j = 0; j < 8; ++j) Ss[c * LS + g * 8 + j] = sv[j];
        }
        __syncthreads();

        // per-kv-chunk stable softmax (zeros from sparsification participate)
        if (tid < 64) {
            int c = tid;
            float mx = -3.402823466e38f;
            for (int e = 0; e < 64; ++e) mx = fmaxf(mx, Ss[c * LS + e]);
            float sum = 0.f;
            for (int e = 0; e < 64; ++e) sum += __expf(Ss[c * LS + e] - mx);
            float inv = 1.f / fmaxf(sum, 1e-6f);
            nrm[c] += sum * inv;
            for (int e = 0; e < 64; ++e)
                Ps[c * LP + e] = (__bf16)(__expf(Ss[c * LS + e] - mx) * inv);
        }
        __syncthreads();

        // out += p @ v   (K = 64, 2 steps; each wave owns 2 out tiles)
        {
            int koA = (lane < 16) ? 0 : 8;
            int koB = (lane < 16) ? 0 : 8;
            int rowp = rq * 16 + lr;
            #pragma unroll
            for (int kk = 0; kk < 2; ++kk) {
                v16bf pa = mkfrag(*(const uint4*)&Ps[rowp * LP + kk * 32 + koA],
                                  *(const uint4*)&Ps[rowp * LP + kk * 32 + koA + 16]);
                #pragma unroll
                for (int t2 = 0; t2 < 2; ++t2) {
                    int d = ((wave & 3) * 2 + t2) * 16 + lr;
                    v16bf vb = mkfrag(*(const uint4*)&Vt[d * LV + kk * 32 + koB],
                                      *(const uint4*)&Vt[d * LV + kk * 32 + koB + 16]);
                    oacc[t2] = __builtin_amdgcn_wmma_f32_16x16x32_bf16(
                        false, pa, false, vb, (short)0, oacc[t2], false, false);
                }
            }
        }
        __syncthreads();
    }

    // normalize + nan_to_num + scatter back to token-major [B*T, H*D]
    #pragma unroll
    for (int t2 = 0; t2 < 2; ++t2) {
        int d = ((wave & 3) * 2 + t2) * 16 + lr;
        #pragma unroll
        for (int r = 0; r < 8; ++r) {
            int c = rq * 16 + ((lane < 16) ? r : r + 8);
            float val = oacc[t2][r] / fmaxf(nrm[c], 1e-6f);
            val = n2n(val);
            attn[(b * T_ + n * CS + c) * C_ + h * DH + d] = (__bf16)val;
        }
    }
}

// --------------------------------------------------------------------- launcher
extern "C" void kernel_launch(void* const* d_in, const int* in_sizes, int n_in,
                              void* d_out, int out_size, void* d_ws, size_t ws_size,
                              hipStream_t stream) {
    (void)in_sizes; (void)n_in; (void)out_size; (void)ws_size;
    const float* x     = (const float*)d_in[0];
    const float* imp   = (const float*)d_in[1];
    const float* temps = (const float*)d_in[2];
    const float* Wq    = (const float*)d_in[3];
    const float* bq    = (const float*)d_in[4];
    const float* Wk    = (const float*)d_in[5];
    const float* bk    = (const float*)d_in[6];
    const float* Wv    = (const float*)d_in[7];
    const float* bv    = (const float*)d_in[8];
    const float* Wo    = (const float*)d_in[9];
    const float* bo    = (const float*)d_in[10];
    float* out = (float*)d_out;

    char* ws = (char*)d_ws;
    size_t off = 0;
    auto alloc = [&](size_t bytes) -> void* {
        void* p = ws + off;
        off += (bytes + 255) & ~(size_t)255;
        return p;
    };
    __bf16* xb  = (__bf16*)alloc((size_t)M_ * C_ * 2);
    __bf16* WqT = (__bf16*)alloc((size_t)C_ * C_ * 2);
    __bf16* WkT = (__bf16*)alloc((size_t)C_ * C_ * 2);
    __bf16* WvT = (__bf16*)alloc((size_t)C_ * C_ * 2);
    __bf16* WoT = (__bf16*)alloc((size_t)C_ * C_ * 2);
    __bf16* qb  = (__bf16*)alloc((size_t)M_ * C_ * 2);
    __bf16* kb  = (__bf16*)alloc((size_t)M_ * C_ * 2);
    __bf16* vb  = (__bf16*)alloc((size_t)M_ * C_ * 2);
    __bf16* ab  = (__bf16*)alloc((size_t)M_ * C_ * 2);

    cast_bf16_kernel<<<2048, 256, 0, stream>>>(x, xb, M_ * C_);

    dim3 tg(C_ / 32, C_ / 32);
    transpose_cast_kernel<<<tg, 256, 0, stream>>>(Wq, WqT);
    transpose_cast_kernel<<<tg, 256, 0, stream>>>(Wk, WkT);
    transpose_cast_kernel<<<tg, 256, 0, stream>>>(Wv, WvT);
    transpose_cast_kernel<<<tg, 256, 0, stream>>>(Wo, WoT);

    dim3 gg(M_ / 128, C_ / 128);
    gemm_bf16_kernel<true,  true ><<<gg, 256, 0, stream>>>(xb, WqT, bq, qb);
    gemm_bf16_kernel<true,  false><<<gg, 256, 0, stream>>>(xb, WkT, bk, kb);
    gemm_bf16_kernel<true,  false><<<gg, 256, 0, stream>>>(xb, WvT, bv, vb);

    attn_kernel<<<B_ * H * NCH, 512, 0, stream>>>(qb, kb, vb, imp, temps, ab);

    gemm_bf16_kernel<false, false><<<gg, 256, 0, stream>>>(ab, WoT, bo, out);
}
